// PropagationNetParamFree_71811853189814
// MI455X (gfx1250) — matrix operand order repaired and checked by reading, not compiled, not run on hardware
//
#include <hip/hip_runtime.h>
#include <hip/hip_bf16.h>

typedef __attribute__((ext_vector_type(16))) _Float16 v16h;
typedef __attribute__((ext_vector_type(8)))  _Float16 v8h;
typedef __attribute__((ext_vector_type(8)))  float    v8f;
typedef __attribute__((ext_vector_type(4)))  unsigned int v4u;
typedef __attribute__((ext_vector_type(8)))  int      v8i;
typedef __attribute__((ext_vector_type(4)))  int      v4i;

#define NN 8192   // nodes
#define DD 128    // feature dim
#define KC 64     // K chunk per TDM tile (= 2 WMMAs of K=32)
#define NITER (NN / KC)

// ---------------------------------------------------------------------------
// Convert f32 row-major features [N,D] -> f16 transposed featT [D,N]
// so B fragment halves (8 consecutive K for a fixed column) are contiguous.
// ---------------------------------------------------------------------------
__global__ void prop_convert_transpose(const float* __restrict__ f,
                                       _Float16* __restrict__ ft) {
    int idx = blockIdx.x * blockDim.x + threadIdx.x;   // over N*D
    int n = idx >> 7;          // / DD
    int c = idx & (DD - 1);    // % DD
    ft[(size_t)c * NN + n] = (_Float16)f[idx];
}

// ---------------------------------------------------------------------------
// TDM: DMA one 16x64 int32 adjacency tile (row-major, stride NN) into LDS.
// D# per CDNA5 ISA ch.8. Issued by ONE wave only (TDM ignores EXEC);
// tracked with TENSORcnt. 6-arg builtin on this toolchain:
//   (uint32x4 g0, int32x8 g1, int32x4, int32x4, int32x8, i32 cpol)
// ---------------------------------------------------------------------------
#if __has_builtin(__builtin_amdgcn_tensor_load_to_lds)
#define HAVE_TDM 1
__device__ __forceinline__ void tdm_load_tile(unsigned lds_off,
                                              const int* gptr) {
    unsigned long long ga = (unsigned long long)(size_t)gptr;
    v4u g0 = { 1u,                                   // count=1, user mode
               lds_off,                              // lds_addr (bytes)
               (unsigned)ga,                         // global_addr[31:0]
               ((unsigned)(ga >> 32) & 0x01FFFFFFu)  // global_addr[56:32]
                   | 0x80000000u };                  // type=2 ("image")
    v8i g1 = { (int)0x00020000,        // data_size=2 -> 4 bytes; no pad/iter
               (int)(0x2000u << 16),   // tensor_dim0 = 8192 (low16 @ [31:16])
               (int)(0x2000u << 16),   // dim0 hi=0 | tensor_dim1 low16 = 8192
               (int)((unsigned)KC << 16), // dim1 hi=0 | tile_dim0 = 64
               16,                     // tile_dim1 = 16 | tile_dim2 = 0
               8192,                   // tensor_dim0_stride low32
               0, 0 };                 // stride0 hi / stride1 (unused, 2D)
    v4i z4 = { 0, 0, 0, 0 };
    v8i z8 = { 0, 0, 0, 0, 0, 0, 0, 0 };
    __builtin_amdgcn_tensor_load_to_lds(g0, g1, z4, z4, z8, 0);
}
#else
#define HAVE_TDM 0
#endif

// ---------------------------------------------------------------------------
// One propagation layer:
//   prop[i,:] = sum_k adj[i,k] * feat[k,:]   (f16 WMMA, f32 accumulate)
//   deg[i]    = sum_k adj[i,k]
//   out[i,:]  = deg>0 ? prop/deg : feat_in_f32[i,:]
// Block: 256 threads = 8 wave32; wave w -> one 16x16 C tile (cols 16w..).
// Workgroup: 16 output rows. Pipeline per K-tile (2 barriers):
//   [wave0: TDM next tile -> int LDS buf, s_wait_tensorcnt] barrier
//   [all: int tile -> f16 tile (bit trick) + degree partials] barrier
//   [all: ds_load_b128 A frags + global B frags -> 2x WMMA]
// ---------------------------------------------------------------------------
__global__ void __launch_bounds__(256)
prop_layer_kernel(const int* __restrict__ adj,        // [NN, NN] 0/1
                  const _Float16* __restrict__ featT, // [DD, NN] f16
                  const float* __restrict__ featIn,   // [NN, DD] f32 (fallback)
                  float* __restrict__ out) {          // [NN, DD] f32
    __shared__ __attribute__((aligned(16))) int      bufInt[2][16 * KC]; // 2x4KB
    __shared__ __attribute__((aligned(16))) _Float16 bufH[2][16 * KC];   // 2x2KB
    __shared__ int degp[256];
    __shared__ int degs[16];

    const int t    = threadIdx.x;
    const int lane = t & 31;
    const int wv   = t >> 5;               // 8 waves
    const int rowg0 = blockIdx.x * 16;

    // scalar (SGPR) wave id -> guaranteed s_cbranch around TDM issue
    const int waveid = __builtin_amdgcn_readfirstlane(wv);

    // convert staging: thread t owns 4 ints of row t>>4, cols (t&15)*4..+3
    const int srow = t >> 4;
    const int scol = (t & 15) * 4;

    // B fragment source: lane column + lane K-half
    const int colg = wv * 16 + (lane & 15);
    const _Float16* __restrict__ bcol =
        featT + (size_t)colg * NN + (lane >> 4) * 8;

    // A fragment (16-bit A 16x32 layout) within the 16x64 f16 tile:
    //   row = lane&15; lanes 0-15: K {0..7,16..23}; lanes 16-31: +8
    const int afragoff = (lane & 15) * KC + (lane >> 4) * 8;

    const int* __restrict__ atile = adj + (size_t)rowg0 * NN;

#if HAVE_TDM
    const unsigned ldsbase = (unsigned)(size_t)(void*)&bufInt[0][0];
    if (waveid == 0) {
        tdm_load_tile(ldsbase, atile);                 // prime buffer 0
    }
#endif

    v8f c = {};
    int degacc = 0;

    for (int i = 0; i < NITER; ++i) {
#if HAVE_TDM
        if (waveid == 0) {
            if (i + 1 < NITER) {   // prefetch next tile into other buffer
                tdm_load_tile(ldsbase + (unsigned)(((i + 1) & 1) * (16 * KC * 4)),
                              atile + (size_t)(i + 1) * KC);
                __builtin_amdgcn_s_wait_tensorcnt(1);  // tile i landed
            } else {
                __builtin_amdgcn_s_wait_tensorcnt(0);
            }
        }
#else
        {   // manual staging fallback: coalesced int4 copy into LDS
            int4 av = *(const int4*)(atile + (size_t)srow * NN + i * KC + scol);
            *(int4*)&bufInt[i & 1][srow * KC + scol] = av;
        }
#endif
        __syncthreads();   // int tile i visible to all waves

        // ---- convert int 0/1 -> f16 tile ONCE per tile + degree partials ---
        // f16(1.0) = 0x3C00 -> pack pair: bits((x + (y<<16)) * 0x3C00)
        {
            const int* __restrict__ ib = &bufInt[i & 1][0];
            int4 q = *(const int4*)(ib + srow * KC + scol);
            degacc += q.x + q.y + q.z + q.w;
            unsigned p0 = (unsigned)(q.x + (q.y << 16)) * 0x3C00u;
            unsigned p1 = (unsigned)(q.z + (q.w << 16)) * 0x3C00u;
            uint2 pk = { p0, p1 };
            *(uint2*)((unsigned*)&bufH[i & 1][0] + srow * (KC / 2) + (t & 15) * 2) = pk;
        }
        __syncthreads();   // f16 tile ready

        // ---- two K=32 WMMAs per tile ----
        const _Float16* __restrict__ hb = &bufH[i & 1][0];
        #pragma unroll
        for (int s = 0; s < 2; ++s) {
            const _Float16* __restrict__ ap = hb + afragoff + s * 32;
            v8h alo = *(const v8h*)(ap);        // K 0..7   (or 8..15)
            v8h ahi = *(const v8h*)(ap + 16);   // K 16..23 (or 24..31)
            v16h a = __builtin_shufflevector(alo, ahi,
                     0,1,2,3,4,5,6,7,8,9,10,11,12,13,14,15);

            const _Float16* bp = bcol + i * KC + s * 32;
            v8h blo = *(const v8h*)(bp);
            v8h bhi = *(const v8h*)(bp + 16);
            v16h b = __builtin_shufflevector(blo, bhi,
                     0,1,2,3,4,5,6,7,8,9,10,11,12,13,14,15);

            c = __builtin_amdgcn_wmma_f32_16x16x32_f16(
                    /*neg_a=*/false, a, /*neg_b=*/false, b,
                    /*c_mod=*/(short)0, c, /*reuse_a=*/false, /*reuse_b=*/false);
        }
        // next iteration's barrier A separates these reads from the refill
    }

    // ---- reduce per-row degree (thread t owns tile row t>>4) ----
    degp[t] = degacc;
    __syncthreads();
    if (t < 16) {
        int s = 0;
        #pragma unroll
        for (int k = 0; k < 16; ++k) s += degp[t * 16 + k];
        degs[t] = s;
    }
    __syncthreads();

    // ---- epilogue: C layout -> VGPR v holds M = v + 8*(lane>=16), N=lane&15
    const int rhalf = (lane >> 4) * 8;
    #pragma unroll
    for (int v = 0; v < 8; ++v) {
        int rloc = v + rhalf;
        int rg   = rowg0 + rloc;
        int dg   = degs[rloc];
        float val;
        if (dg > 0) {
            val = c[v] * (1.0f / (float)dg);
        } else {
            val = featIn[(size_t)rg * DD + colg];
        }
        out[(size_t)rg * DD + colg] = val;
    }
}

// ---------------------------------------------------------------------------
// Harness entry. Inputs: d_in[0]=features f32 [8192,128],
//                        d_in[1]=adj_lst int32 [2,8192,8192].
// Output: f32 [8192,128]. Workspace: 2MB f16 featT + 4MB f32 intermediate.
// ---------------------------------------------------------------------------
extern "C" void kernel_launch(void* const* d_in, const int* in_sizes, int n_in,
                              void* d_out, int out_size, void* d_ws, size_t ws_size,
                              hipStream_t stream) {
    const float* feat = (const float*)d_in[0];
    const int*   adj  = (const int*)d_in[1];
    float*       outp = (float*)d_out;

    char* ws = (char*)d_ws;
    _Float16* featT = (_Float16*)ws;                                      // 2 MB
    float*    feat1 = (float*)(ws + (size_t)NN * DD * sizeof(_Float16));  // 4 MB

    dim3 cvtGrid((NN * DD) / 256), cvtBlk(256);
    dim3 gemmGrid(NN / 16), gemmBlk(256);

    // Layer 0: adj[0] x feat -> feat1
    prop_convert_transpose<<<cvtGrid, cvtBlk, 0, stream>>>(feat, featT);
    prop_layer_kernel<<<gemmGrid, gemmBlk, 0, stream>>>(adj, featT, feat, feat1);

    // Layer 1: adj[1] x feat1 -> out
    prop_convert_transpose<<<cvtGrid, cvtBlk, 0, stream>>>(feat1, featT);
    prop_layer_kernel<<<gemmGrid, gemmBlk, 0, stream>>>(
        adj + (size_t)NN * NN, featT, feat1, outp);
}
